// graphsage_27273042329874
// MI455X (gfx1250) — matrix operand order repaired and checked by reading, not compiled, not run on hardware
//
#include <hip/hip_runtime.h>
#include <hip/hip_bf16.h>

// ---------------------------------------------------------------------------
// Problem constants (from reference)
// ---------------------------------------------------------------------------
static constexpr int S1 = 73216;   // layer-1 src rows
static constexpr int D1 = 2816;    // layer-1 dst rows
static constexpr int S2 = 2816;    // layer-2 src rows
static constexpr int D2 = 256;     // layer-2 dst rows
static constexpr int F  = 256;     // feature width / internal width
static constexpr int KSPLIT = 8;   // split-K factor for the diffusion GEMMs

// ---------------------------------------------------------------------------
// CDNA5 WMMA types
// ---------------------------------------------------------------------------
typedef __attribute__((ext_vector_type(16))) __bf16          v16bf;
typedef __attribute__((ext_vector_type(8)))  float           v8f;
typedef __attribute__((ext_vector_type(8)))  unsigned short  u16x8;
typedef __attribute__((ext_vector_type(4)))  float           f32x4;
typedef __attribute__((ext_vector_type(4)))  __bf16          bf16x4;

union frag_u {
    struct { u16x8 lo, hi; } h;
    v16bf v;
};

// native HW bf16 conversion (RTNE) — lets the backend use packed v_cvt ops
__device__ __forceinline__ unsigned short f2bf(float x) {
    union { __bf16 b; unsigned short u; } c;
    c.b = (__bf16)x;
    return c.u;
}

// ---------------------------------------------------------------------------
// small helper kernels
// ---------------------------------------------------------------------------
__global__ void cvt_f32_bf16(const float* __restrict__ src,
                             unsigned short* __restrict__ dst, int n) {
    int i = blockIdx.x * blockDim.x + threadIdx.x;
    if (i < n) dst[i] = f2bf(src[i]);
}

// gather with double indirection from fp32 features -> bf16
__global__ void gather2_f32_bf16(const float* __restrict__ feat,
                                 const int* __restrict__ outer,
                                 const int* __restrict__ inner,
                                 unsigned short* __restrict__ dst,
                                 int stride, int coloff) {
    int j = blockIdx.x, f = threadIdx.x;
    int row = outer[inner[j]];
    dst[(long)j * stride + coloff + f] = f2bf(feat[(long)row * F + f]);
}

// gather bf16 -> bf16 (single indirection)
__global__ void gather_bf16(const unsigned short* __restrict__ src,
                            const int* __restrict__ idx,
                            unsigned short* __restrict__ dst,
                            int stride, int coloff) {
    int j = blockIdx.x, f = threadIdx.x;
    int row = idx[j];
    dst[(long)j * stride + coloff + f] = src[(long)row * F + f];
}

// sum split-K fp32 partials -> bf16 (into a strided destination)
__global__ void reduce_partials_bf16(const float* __restrict__ part,
                                     unsigned short* __restrict__ dst,
                                     int M, int ldd, int coloff, int zcount) {
    int m = blockIdx.x, n = threadIdx.x;
    float s = 0.f;
#pragma unroll
    for (int z = 0; z < KSPLIT; ++z)
        if (z < zcount) s += part[((long)z * M + m) * F + n];
    dst[(long)m * ldd + coloff + n] = f2bf(s);
}

// row-wise L2 normalize: out_bf16[j,:] = x[j,:] * rsqrt(max(sum(x^2), 1e-12))
__global__ void l2norm_bf16(const float* __restrict__ x,
                            unsigned short* __restrict__ out) {
    __shared__ float red[256];
    int j = blockIdx.x, f = threadIdx.x;
    float v = x[(long)j * F + f];
    red[f] = v * v;
    __syncthreads();
    for (int s = 128; s > 0; s >>= 1) {
        if (f < s) red[f] += red[f + s];
        __syncthreads();
    }
    float inv = rsqrtf(fmaxf(red[0], 1e-12f));
    out[(long)j * F + f] = f2bf(v * inv);
}

// ---------------------------------------------------------------------------
// Tiled WMMA GEMM with optional split-K:
//   C[M,N] = act(A[M,K] @ B[K,N] + bias)           (ksplit == 1)
//   Cpart[z][M,N] = A[M, Kz] @ B[Kz, N]            (ksplit > 1, fp32 partials)
//
// REQUIRES: M % 128 == 0, K % (32*ksplit) == 0. N may be ragged but N % 8 == 0.
//   A: fp32 (non-temporal stream, converted in-register) or bf16, row-major
//   B: bf16 row-major
//
// Block: 256 threads = 8 waves; block tile 128(M) x 64(N), K-step 32.
// Wave w owns rows w*16..w*16+15 and all 64 cols: one A fragment feeds FOUR
// v_wmma_f32_16x16x32_bf16 per K-step. Staging is branchless and vectorized
// (global_load_b128 -> ds_store_b128 / masked B), software-pipelined so the
// next tile's global loads overlap the current tile's WMMAs.
// ---------------------------------------------------------------------------
template <int A_IS_F32>
__global__ __launch_bounds__(256)
void wmma_gemm(const void* __restrict__ Ap,
               const unsigned short* __restrict__ Bp,
               float* __restrict__ Cf,
               unsigned short* __restrict__ Cbf,
               const float* __restrict__ bias,
               int M, int N, int K, int lda, int ldb, int ldc,
               int relu, int ksplit) {
    __shared__ __align__(16) unsigned short lA[128 * 40];  // [m][k], stride 40
    __shared__ __align__(16) unsigned short lB[64 * 40];   // [n][k], stride 40

    const int tid  = threadIdx.x;
    const int row0 = blockIdx.y * 128;
    const int col0 = blockIdx.x * 64;
    const int wm   = tid >> 5;          // wave id = M sub-tile 0..7
    const int lane = tid & 31;
    const int half = lane >> 4;
    const int mr   = lane & 15;

    // ---- staging assignment ----
    const int am = tid >> 1;            // 0..127  A row within tile
    const int ak = (tid & 1) << 4;      // 0 or 16 (16 elements each)
    const int bk = tid >> 3;            // 0..31   B row (k) within tile
    const int bn = (tid & 7) << 3;      // 0..56   B col start within tile

    // branchless B edge handling: N % 8 == 0, so each 8-seg is fully in or out
    const int  gnb    = col0 + bn;
    const int  bstart = (gnb + 8 <= N) ? gnb : (N - 8);
    u16x8 bmask;
#pragma unroll
    for (int j = 0; j < 8; ++j)
        bmask[j] = (gnb + j < N) ? (unsigned short)0xFFFFu : (unsigned short)0;

    // ---- split-K range ----
    const int tperz = (K >> 5) / ksplit;
    const int t0    = blockIdx.z * tperz;
    const int t1    = t0 + tperz;

    // running global pointers (no per-tile 64-bit multiplies)
    const float*          aF = (const float*)Ap +
                               (long)(row0 + am) * lda + ak + (long)t0 * 32;
    const unsigned short* aH = (const unsigned short*)Ap +
                               (long)(row0 + am) * lda + ak + (long)t0 * 32;
    const unsigned short* bP = Bp + ((long)t0 * 32 + bk) * ldb + bstart;

    // staging registers
    f32x4 ra0 = {}, ra1 = {}, ra2 = {}, ra3 = {};
    u16x8 rh0 = {}, rh1 = {};
    u16x8 rb  = {};

    auto load_tile = [&]() {
        if (A_IS_F32) {
            const f32x4* p = (const f32x4*)aF;
            ra0 = __builtin_nontemporal_load(p);       // streamed once: keep L2
            ra1 = __builtin_nontemporal_load(p + 1);   // for the reused B panel
            ra2 = __builtin_nontemporal_load(p + 2);
            ra3 = __builtin_nontemporal_load(p + 3);
        } else {
            rh0 = *(const u16x8*)(aH);
            rh1 = *(const u16x8*)(aH + 8);
        }
        rb = *(const u16x8*)(bP) & bmask;              // branchless masked load
        aF += 32;
        aH += 32;
        bP += (long)32 * ldb;
    };

    auto store_tile = [&]() {
        u16x8 ac0, ac1;
        if (A_IS_F32) {
            union { struct { bf16x4 a, b; } s; u16x8 u; } p0, p1;
            p0.s.a = __builtin_convertvector(ra0, bf16x4);  // packed HW cvt
            p0.s.b = __builtin_convertvector(ra1, bf16x4);
            p1.s.a = __builtin_convertvector(ra2, bf16x4);
            p1.s.b = __builtin_convertvector(ra3, bf16x4);
            ac0 = p0.u;
            ac1 = p1.u;
        } else {
            ac0 = rh0;
            ac1 = rh1;
        }
        *(u16x8*)&lA[am * 40 + ak]     = ac0;          // ds_store_b128
        *(u16x8*)&lA[am * 40 + ak + 8] = ac1;          // ds_store_b128
#pragma unroll
        for (int j = 0; j < 8; ++j)                    // transpose scatter
            lB[(bn + j) * 40 + bk] = rb[j];
    };

    v8f acc[4] = {{}, {}, {}, {}};

    load_tile();
#pragma unroll 1
    for (int kt = t0; kt < t1; ++kt) {
        store_tile();
        __syncthreads();
        if (kt + 1 < t1) load_tile();                  // prefetch next K-tile

        // ---- fragments (ISA 7.12.2 layouts) ----
        // A: lane holds row M=mr, K in {8*half+0..7} U {16+8*half+0..7}
        const unsigned short* ap = &lA[(wm * 16 + mr) * 40 + half * 8];
        frag_u fa;
        fa.h.lo = *(const u16x8*)(ap);
        fa.h.hi = *(const u16x8*)(ap + 16);
        // B: lane holds col N=mr, K in {8*half+0..7} U {16+8*half+0..7}
        frag_u fb[4];
#pragma unroll
        for (int j = 0; j < 4; ++j) {
            const unsigned short* bp = &lB[(j * 16 + mr) * 40 + half * 8];
            fb[j].h.lo = *(const u16x8*)(bp);
            fb[j].h.hi = *(const u16x8*)(bp + 16);
        }
#pragma unroll
        for (int j = 0; j < 4; ++j)
            acc[j] = __builtin_amdgcn_wmma_f32_16x16x32_bf16(
                false, fa.v, false, fb[j].v, (short)0, acc[j], false, false);
        __syncthreads();
    }

    // ---- epilogue: C lane layout: n = mr, rows r + 8*half ----
    float* cf = Cf ? Cf + (size_t)blockIdx.z * M * ldc : nullptr;
#pragma unroll
    for (int j = 0; j < 4; ++j) {
        int  n   = col0 + j * 16 + mr;
        bool nok = n < N;
        float bv = (bias && nok) ? bias[n] : 0.f;
#pragma unroll
        for (int r = 0; r < 8; ++r) {
            int   m = row0 + wm * 16 + r + half * 8;   // always < M (M%128==0)
            float v = acc[j][r] + bv;
            if (relu) v = fmaxf(v, 0.f);
            if (nok) {
                if (cf)  cf [(long)m * ldc + n] = v;
                if (Cbf) Cbf[(long)m * ldc + n] = f2bf(v);
            }
        }
    }
}

// ---------------------------------------------------------------------------
// Host-side orchestration
// ---------------------------------------------------------------------------
extern "C" void kernel_launch(void* const* d_in, const int* in_sizes, int n_in,
                              void* d_out, int out_size, void* d_ws, size_t ws_size,
                              hipStream_t stream) {
    (void)in_sizes; (void)n_in; (void)out_size; (void)ws_size;

    const int*   src_nodes   = (const int*)  d_in[0];
    const int*   dstsrc2src1 = (const int*)  d_in[1];
    const int*   dstsrc2dst1 = (const int*)  d_in[2];
    const float* dif_mat1    = (const float*)d_in[3];
    const int*   dstsrc2src2 = (const int*)  d_in[4];
    const int*   dstsrc2dst2 = (const int*)  d_in[5];
    const float* dif_mat2    = (const float*)d_in[6];
    const float* features    = (const float*)d_in[7];
    const float* w1          = (const float*)d_in[8];
    const float* w2          = (const float*)d_in[9];
    const float* W3          = (const float*)d_in[10];
    const float* b3          = (const float*)d_in[11];
    const float* W4          = (const float*)d_in[12];
    const float* b4          = (const float*)d_in[13];
    const float* W5          = (const float*)d_in[14];
    const float* b5          = (const float*)d_in[15];
    float* out = (float*)d_out;

    // workspace carve-out (256B aligned)
    char* ws = (char*)d_ws;
    size_t off = 0;
    auto carve = [&](size_t bytes) -> void* {
        off = (off + 255) & ~(size_t)255;
        void* p = ws + off;
        off += bytes;
        return p;
    };
    unsigned short* G1bf     = (unsigned short*)carve((size_t)S1 * F * 2);  // 37.5 MB
    float*          partials = (float*)        carve((size_t)KSPLIT * D1 * F * 4); // 23 MB
    unsigned short* concat1  = (unsigned short*)carve((size_t)D1 * 2 * F * 2);
    unsigned short* h1bf     = (unsigned short*)carve((size_t)D1 * F * 2);
    unsigned short* G2bf     = (unsigned short*)carve((size_t)S2 * F * 2);
    unsigned short* concat2  = (unsigned short*)carve((size_t)D2 * 2 * F * 2);
    float*          x2f      = (float*)        carve((size_t)D2 * F * 4);
    unsigned short* embbf    = (unsigned short*)carve((size_t)D2 * F * 2);
    unsigned short* h3bf     = (unsigned short*)carve((size_t)D2 * 64 * 2);
    unsigned short* h4bf     = (unsigned short*)carve((size_t)D2 * 64 * 2);
    unsigned short* w1bf     = (unsigned short*)carve((size_t)512 * 256 * 2);
    unsigned short* w2bf     = (unsigned short*)carve((size_t)512 * 256 * 2);
    unsigned short* W3bf     = (unsigned short*)carve((size_t)256 * 64 * 2);
    unsigned short* W4bf     = (unsigned short*)carve((size_t)64 * 64 * 2);
    unsigned short* W5bf     = (unsigned short*)carve((size_t)64 * 8 * 2);

    // ---- weight conversions ----
    cvt_f32_bf16<<<(512 * 256 + 255) / 256, 256, 0, stream>>>(w1, w1bf, 512 * 256);
    cvt_f32_bf16<<<(512 * 256 + 255) / 256, 256, 0, stream>>>(w2, w2bf, 512 * 256);
    cvt_f32_bf16<<<(256 * 64 + 255) / 256, 256, 0, stream>>>(W3, W3bf, 256 * 64);
    cvt_f32_bf16<<<(64 * 64 + 255) / 256, 256, 0, stream>>>(W4, W4bf, 64 * 64);
    cvt_f32_bf16<<<(64 * 8 + 255) / 256, 256, 0, stream>>>(W5, W5bf, 64 * 8);

    // ---- layer 1 gathers (fuse features[src_nodes[...]]) ----
    gather2_f32_bf16<<<S1, 256, 0, stream>>>(features, src_nodes, dstsrc2src1,
                                             G1bf, F, 0);
    gather2_f32_bf16<<<D1, 256, 0, stream>>>(features, src_nodes, dstsrc2dst1,
                                             concat1, 2 * F, F);

    // ---- agg1 = dif_mat1 @ G1 : split-K partials, reduce -> concat1 left ----
    {
        dim3 g(F / 64, D1 / 128, KSPLIT);   // (4, 22, 8) = 704 blocks
        wmma_gemm<1><<<g, 256, 0, stream>>>(dif_mat1, G1bf, partials, nullptr,
                                            nullptr, D1, F, S1, S1, F, F, 0, KSPLIT);
        reduce_partials_bf16<<<D1, 256, 0, stream>>>(partials, concat1,
                                                     D1, 2 * F, 0, KSPLIT);
    }
    // ---- h1 = concat1 @ w1 (no activation) ----
    {
        dim3 g(F / 64, D1 / 128);
        wmma_gemm<0><<<g, 256, 0, stream>>>(concat1, w1bf, nullptr, h1bf,
                                            nullptr, D1, F, 2 * F, 2 * F, F, F, 0, 1);
    }

    // ---- layer 2 gathers on h1 ----
    gather_bf16<<<S2, 256, 0, stream>>>(h1bf, dstsrc2src2, G2bf, F, 0);
    gather_bf16<<<D2, 256, 0, stream>>>(h1bf, dstsrc2dst2, concat2, 2 * F, F);

    // ---- agg2 = dif_mat2 @ G2 : split-K, reduce -> concat2 left ----
    {
        dim3 g(F / 64, D2 / 128, KSPLIT);   // (4, 2, 8) = 64 blocks
        wmma_gemm<1><<<g, 256, 0, stream>>>(dif_mat2, G2bf, partials, nullptr,
                                            nullptr, D2, F, S2, S2, F, F, 0, KSPLIT);
        reduce_partials_bf16<<<D2, 256, 0, stream>>>(partials, concat2,
                                                     D2, 2 * F, 0, KSPLIT);
    }
    // ---- x2 = relu(concat2 @ w2) (fp32 for normalize) ----
    {
        dim3 g(F / 64, D2 / 128);
        wmma_gemm<0><<<g, 256, 0, stream>>>(concat2, w2bf, x2f, nullptr,
                                            nullptr, D2, F, 2 * F, 2 * F, F, F, 1, 1);
    }

    // ---- L2 normalize rows -> bf16 embeddings ----
    l2norm_bf16<<<D2, 256, 0, stream>>>(x2f, embbf);

    // ---- MLP head ----
    {
        dim3 g(1, D2 / 128);
        wmma_gemm<0><<<g, 256, 0, stream>>>(embbf, W3bf, nullptr, h3bf,
                                            b3, D2, 64, 256, 256, 64, 64, 1, 1);
        wmma_gemm<0><<<g, 256, 0, stream>>>(h3bf, W4bf, nullptr, h4bf,
                                            b4, D2, 64, 64, 64, 64, 64, 1, 1);
        wmma_gemm<0><<<g, 256, 0, stream>>>(h4bf, W5bf, out, nullptr,
                                            b5, D2, 8, 64, 64, 8, 8, 0, 1);
    }
}
